// MLP3D_final_74758200754254
// MI455X (gfx1250) — matrix operand order repaired and checked by reading, not compiled, not run on hardware
//
#include <hip/hip_runtime.h>

typedef __attribute__((ext_vector_type(16))) _Float16 v16h;
typedef __attribute__((ext_vector_type(8)))  _Float16 v8h;
typedef __attribute__((ext_vector_type(8)))  float    v8f;

#define HIDDEN   256
#define NTILES_H 16     // 256 / 16 output tiles
#define KBLK_H   8      // 256 / 32 k-blocks
#define WAVES    8      // 8 wave32 per workgroup (256 threads)

static __device__ __forceinline__ v16h concat16(v8h lo, v8h hh) {
  return __builtin_shufflevector(lo, hh, 0,1,2,3,4,5,6,7,8,9,10,11,12,13,14,15);
}

static __device__ __forceinline__ v8f wmma16(v16h a, v16h b, v8f c) {
  // D = A(16x32 f16) x B(32x16 f16) + C(16x16 f32)
  return __builtin_amdgcn_wmma_f32_16x16x32_f16(false, a, false, b, (short)0, c, false, false);
}

// ---------------------------------------------------------------------------
// Pack f32 weights -> f16 B-operand block layout: [kb][nt][lane][16 halves]
// lane%16 = N column in tile; with h = lane/16, half i of a lane holds
// K = kb*32 + 8h + (i&7) + 16*(i>>3).  K rows beyond krows are zeroed (pads W0 63 -> 64).
// ---------------------------------------------------------------------------
__global__ void pack_weights_kernel(const float* __restrict__ W0,
                                    const float* __restrict__ W1,
                                    const float* __restrict__ W2,
                                    _Float16* __restrict__ w0p,
                                    _Float16* __restrict__ w1p,
                                    _Float16* __restrict__ w2p)
{
  int id = blockIdx.x * blockDim.x + threadIdx.x;
  const float* src; _Float16* dst; int e; int krows;
  if (id < 16384)              { src = W0; dst = w0p; e = id;          krows = 63;  }
  else if (id < 16384 + 65536) { src = W1; dst = w1p; e = id - 16384;  krows = 256; }
  else if (id < 16384 +131072) { src = W2; dst = w2p; e = id - 81920;  krows = 256; }
  else return;
  int i    = e & 15;
  int lane = (e >> 4) & 31;
  int nt   = (e >> 9) & 15;
  int kb   = e >> 13;
  int K = kb * 32 + (lane >> 4) * 8 + (i & 7) + (i >> 3) * 16;
  int N = nt * 16 + (lane & 15);
  float v = (K < krows) ? src[K * HIDDEN + N] : 0.0f;
  dst[e] = (_Float16)v;
}

// ---------------------------------------------------------------------------
// NeRF positional embedding, scattered directly into WMMA A-operand layout:
// embA[((tile*2 + kb)*32 + lane)*16 + idx] -> L0's A operand is two
// contiguous 32 B loads per lane.  One thread per element.
// ---------------------------------------------------------------------------
__global__ void embed_pack_kernel(const float* __restrict__ coords,
                                  _Float16* __restrict__ embA, int npts)
{
  int id = blockIdx.x * blockDim.x + threadIdx.x;
  if (id >= npts * 64) return;
  const int pt = id >> 6;
  const int e  = id & 63;
  const float* c = coords + pt * 3;
  float v;
  if (e < 3) {
    v = c[e];
  } else if (e < 63) {
    const int j = e - 3;
    const int f = j / 6;
    const int r = j % 6;
    const float xv = c[r % 3] * (float)(1 << f);
    v = (r < 3) ? __sinf(xv) : __cosf(xv);
  } else {
    v = 0.0f;                       // pad col 63 (matches zeroed W0 row 63)
  }
  const int t  = pt >> 4;
  const int m  = pt & 15;
  const int kb = e >> 5;
  const int kk = e & 31;
  const int hi = (kk >> 3) & 1;
  const int idx = ((kk >> 4) << 3) | (kk & 7);
  embA[(((t * 2 + kb) * 32) + (hi * 16 + m)) * 16 + idx] = (_Float16)v;
}

// ---------------------------------------------------------------------------
// Fused MLP: one wave = one 16-point tile.
// ---------------------------------------------------------------------------
__global__ __launch_bounds__(256, 1)
void mlp3d_fused_kernel(const float* __restrict__ b0,
                        const float* __restrict__ b1,
                        const float* __restrict__ b2,
                        const float* __restrict__ Wocc,
                        const float* __restrict__ bocc,
                        const float* __restrict__ Wc,
                        const float* __restrict__ bc,
                        const _Float16* __restrict__ embA,
                        const _Float16* __restrict__ w0p,
                        const _Float16* __restrict__ w1p,
                        const _Float16* __restrict__ w2p,
                        float* __restrict__ out_occ,
                        float* __restrict__ out_pc,
                        int ntiles)
{
  // 128K W1 + 32K W0 + 2 x 64K per-wave activation buffers = 288 KiB LDS
  __shared__ _Float16 __attribute__((aligned(32))) lds_w1[KBLK_H * 16 * 32 * 16];
  __shared__ _Float16 __attribute__((aligned(32))) lds_w0[2 * 16 * 32 * 16];
  __shared__ _Float16 __attribute__((aligned(32))) bufX[WAVES][16 * HIDDEN];
  __shared__ _Float16 __attribute__((aligned(32))) bufH[WAVES][16 * HIDDEN];

  // warm L2 with W2 while we stage W0/W1 (global_prefetch_b8)
  for (int i = threadIdx.x; i < 1024; i += 256)
    __builtin_prefetch((const char*)w2p + i * 128, 0, 0);

  {
    const float4* __restrict__ s1 = (const float4*)w1p;
    float4* d1 = (float4*)lds_w1;
    for (int i = threadIdx.x; i < (KBLK_H * 16 * 32 * 16) / 8; i += 256) d1[i] = s1[i];
    const float4* __restrict__ s0 = (const float4*)w0p;
    float4* d0 = (float4*)lds_w0;
    for (int i = threadIdx.x; i < (2 * 16 * 32 * 16) / 8; i += 256) d0[i] = s0[i];
  }
  __syncthreads();

  const int wave = threadIdx.x >> 5;
  const int lane = threadIdx.x & 31;
  const int l16  = lane & 15;
  const int hi   = lane >> 4;

  _Float16* X = bufX[wave];
  _Float16* H = bufH[wave];

  // loop-invariant per-kb B base pointers: the per-nt offset nt*512 halves
  // (= nt*1024 B, max 15360) folds into the load-offset immediates.
  const _Float16* w1b[KBLK_H];
  const _Float16* w2b[KBLK_H];
  #pragma unroll
  for (int kb = 0; kb < KBLK_H; ++kb) {
    w1b[kb] = lds_w1 + lane * 16 + kb * 16 * 512;
    w2b[kb] = w2p    + lane * 16 + kb * 16 * 512;
  }
  const _Float16* w0b[2];
  #pragma unroll
  for (int kb = 0; kb < 2; ++kb)
    w0b[kb] = lds_w0 + lane * 16 + kb * 16 * 512;

  const int gw  = blockIdx.x * WAVES + wave;
  const int gws = gridDim.x * WAVES;

  for (int t = gw; t < ntiles; t += gws) {
    const int row0 = t * 16;

    // ---- A operand for layer 0: precomputed embedding, 2 global 32B loads
    v16h ae[2];
    #pragma unroll
    for (int kb = 0; kb < 2; ++kb)
      ae[kb] = *(const v16h*)(embA + ((t * 2 + kb) * 32 + lane) * 16);

    // ============ layer 0: x = embed @ W0 + b0  (no activation) ============
    {
      v16h bb[2];
      #pragma unroll
      for (int kb = 0; kb < 2; ++kb)
        bb[kb] = *(const v16h*)(w0b[kb]);
      _Float16* dst = X + (8 * hi) * HIDDEN + l16;   // running pointer
      const float* bp = b0 + l16;
      for (int nt = 0; nt < NTILES_H; ++nt) {
        const int ntn = (nt < NTILES_H - 1) ? nt + 1 : nt;
        v8f acc = {};
        #pragma unroll
        for (int kb = 0; kb < 2; ++kb) {
          const v16h bcur = bb[kb];
          bb[kb] = *(const v16h*)(w0b[kb] + ntn * 512);
          acc = wmma16(ae[kb], bcur, acc);
        }
        const float bias = bp[nt * 16];
        #pragma unroll
        for (int v = 0; v < 8; ++v)
          dst[v * HIDDEN] = (_Float16)(acc[v] + bias);
        dst += 16;
      }
    }
    asm volatile("s_wait_dscnt 0" ::: "memory");

    // ---- cache x as A operands (8 k-blocks, 64 VGPRs)
    v16h ax[KBLK_H];
    #pragma unroll
    for (int kb = 0; kb < KBLK_H; ++kb) {
      const _Float16* ap = X + l16 * HIDDEN + kb * 32 + hi * 8;
      ax[kb] = concat16(*(const v8h*)ap, *(const v8h*)(ap + 16));
    }

    // ============ occ trunk layer 1: h1 = relu(x @ W1 + b1) -> H ===========
    {
      v16h bb[KBLK_H];
      #pragma unroll
      for (int kb = 0; kb < KBLK_H; ++kb)
        bb[kb] = *(const v16h*)(w1b[kb]);
      _Float16* dst = H + (8 * hi) * HIDDEN + l16;   // running pointer
      const float* bp = b1 + l16;
      for (int nt = 0; nt < NTILES_H; ++nt) {
        const int ntn = (nt < NTILES_H - 1) ? nt + 1 : nt;
        v8f acc = {};
        #pragma unroll
        for (int kb = 0; kb < KBLK_H; ++kb) {
          const v16h bcur = bb[kb];
          bb[kb] = *(const v16h*)(w1b[kb] + ntn * 512);
          acc = wmma16(ax[kb], bcur, acc);
        }
        const float bias = bp[nt * 16];
        #pragma unroll
        for (int v = 0; v < 8; ++v)
          dst[v * HIDDEN] = (_Float16)fmaxf(acc[v] + bias, 0.0f);
        dst += 16;
      }
    }

    // ============ part branches + class heads (mask => K in [64p,64p+64),
    //              head mask => only output tiles 4p..4p+3 contribute) ======
    #pragma unroll
    for (int pp = 0; pp < 4; ++pp) {
      float cls[8] = {0.f,0.f,0.f,0.f,0.f,0.f,0.f,0.f};
      for (int nt = 4 * pp; nt < 4 * pp + 4; ++nt) {
        v8f acc = {};
        #pragma unroll
        for (int kb = 2 * pp; kb < 2 * pp + 2; ++kb) {
          const v16h b = *(const v16h*)(w1b[kb] + nt * 512);
          acc = wmma16(ax[kb], b, acc);
        }
        const int n = nt * 16 + l16;
        const float bias = b1[n];
        const float wcv  = Wc[pp * HIDDEN + n];
        #pragma unroll
        for (int v = 0; v < 8; ++v)
          cls[v] += fmaxf(acc[v] + bias, 0.0f) * wcv;
      }
      #pragma unroll
      for (int m = 1; m < 16; m <<= 1) {
        #pragma unroll
        for (int v = 0; v < 8; ++v)
          cls[v] += __shfl_xor(cls[v], m, 32);
      }
      if (l16 == 0) {
        const float bcp = bc[pp];
        #pragma unroll
        for (int v = 0; v < 8; ++v)
          out_pc[(row0 + v + 8 * hi) * 4 + pp] = cls[v] + bcp;
      }
    }

    asm volatile("s_wait_dscnt 0" ::: "memory");

    // ---- cache h1 as A operands
    v16h ah[KBLK_H];
    #pragma unroll
    for (int kb = 0; kb < KBLK_H; ++kb) {
      const _Float16* ap = H + l16 * HIDDEN + kb * 32 + hi * 8;
      ah[kb] = concat16(*(const v8h*)ap, *(const v8h*)(ap + 16));
    }

    // ============ layer 2 (W2 from L2, double-buffered) + fused occ head ===
    float occ[8] = {0.f,0.f,0.f,0.f,0.f,0.f,0.f,0.f};
    {
      v16h bb[KBLK_H];
      #pragma unroll
      for (int kb = 0; kb < KBLK_H; ++kb)
        bb[kb] = *(const v16h*)(w2b[kb]);
      const float* b2p = b2 + l16;
      const float* wop = Wocc + l16;
      for (int nt = 0; nt < NTILES_H; ++nt) {
        const int ntn = (nt < NTILES_H - 1) ? nt + 1 : nt;
        v8f acc = {};
        #pragma unroll
        for (int kb = 0; kb < KBLK_H; ++kb) {
          const v16h bcur = bb[kb];
          bb[kb] = *(const v16h*)(w2b[kb] + ntn * 512);
          acc = wmma16(ah[kb], bcur, acc);
        }
        const float bias = b2p[nt * 16];
        const float wv   = wop[nt * 16];
        #pragma unroll
        for (int v = 0; v < 8; ++v)
          occ[v] += fmaxf(acc[v] + bias, 0.0f) * wv;
      }
    }
    #pragma unroll
    for (int m = 1; m < 16; m <<= 1) {
      #pragma unroll
      for (int v = 0; v < 8; ++v)
        occ[v] += __shfl_xor(occ[v], m, 32);
    }
    if (l16 == 0) {
      const float bo = bocc[0];
      #pragma unroll
      for (int v = 0; v < 8; ++v)
        out_occ[row0 + v + 8 * hi] = occ[v] + bo;
    }
  }
}

// ---------------------------------------------------------------------------
extern "C" void kernel_launch(void* const* d_in, const int* in_sizes, int n_in,
                              void* d_out, int out_size, void* d_ws, size_t ws_size,
                              hipStream_t stream) {
  const float* coords = (const float*)d_in[0];
  const float* W0     = (const float*)d_in[1];
  const float* b0     = (const float*)d_in[2];
  const float* W1     = (const float*)d_in[3];
  const float* b1     = (const float*)d_in[4];
  const float* W2     = (const float*)d_in[5];
  const float* b2     = (const float*)d_in[6];
  const float* Wocc   = (const float*)d_in[7];
  const float* bocc   = (const float*)d_in[8];
  const float* Wc     = (const float*)d_in[9];
  const float* bc     = (const float*)d_in[10];

  const int npts   = in_sizes[0] / 3;   // B == 1
  const int ntiles = npts / 16;

  float* out_occ = (float*)d_out;
  float* out_pc  = (float*)d_out + npts;

  _Float16* w0p  = (_Float16*)d_ws;     // 16384 halves  (32 KiB)
  _Float16* w1p  = w0p + 16384;         // 65536 halves  (128 KiB)
  _Float16* w2p  = w1p + 65536;         // 65536 halves  (128 KiB)
  _Float16* embA = w2p + 65536;         // npts*64 halves (16 MiB @ 131072 pts)

  // 1) weights -> f16 WMMA B-operand layout
  pack_weights_kernel<<<576, 256, 0, stream>>>(W0, W1, W2, w0p, w1p, w2p);

  // 2) embedding -> f16 WMMA A-operand layout (one thread per element)
  embed_pack_kernel<<<(npts * 64 + 255) / 256, 256, 0, stream>>>(coords, embA, npts);

  // 3) fused MLP
  int grid = (ntiles + WAVES - 1) / WAVES;
  if (grid > 512) grid = 512;           // 512 WGs -> 2 tiles/wave, weights staged once
  mlp3d_fused_kernel<<<grid, 256, 0, stream>>>(b0, b1, b2, Wocc, bocc, Wc, bc,
                                               embA, w0p, w1p, w2p,
                                               out_occ, out_pc, ntiles);
}
